// GlobalPointer_Chaizi_86681029968034
// MI455X (gfx1250) — matrix operand order repaired and checked by reading, not compiled, not run on hardware
//
#include <hip/hip_runtime.h>
#include <hip/hip_bf16.h>

typedef __attribute__((ext_vector_type(16))) _Float16 v16h;
typedef __attribute__((ext_vector_type(8)))  _Float16 v8h;
typedef __attribute__((ext_vector_type(8)))  float    v8f;

#define ENT     9
#define DIM     64
#define HIDDEN  768
#define NOUT    1152      // ENT * 2 * DIM
#define KTOT    816       // HIDDEN + 48
#define KPAD    832       // 26 * 32
#define SEQ     512
#define BATCH   32
#define BIGV    1000000000000.0f

__device__ __forceinline__ v8h cvt8(float4 a, float4 b) {
    v8h r;
    r[0] = (_Float16)a.x; r[1] = (_Float16)a.y;
    r[2] = (_Float16)a.z; r[3] = (_Float16)a.w;
    r[4] = (_Float16)b.x; r[5] = (_Float16)b.y;
    r[6] = (_Float16)b.z; r[7] = (_Float16)b.w;
    return r;
}

// One 32-deep k-step: wave computes 4 16x16 tiles from LDS-staged A/B.
__device__ __forceinline__ void wmma_step(const _Float16* As, const _Float16* Bs,
                                          int msub, int nsub, int ll, int hi,
                                          v8f acc[4]) {
    // A fragment: row = msub+ll ; lo lanes k {0..7,16..23}, hi lanes {8..15,24..31}
    const _Float16* ap = &As[(msub + ll) * 40 + hi * 8];
    v16h a;
    *((v8h*)&a)       = *(const v8h*)(ap);
    *(((v8h*)&a) + 1) = *(const v8h*)(ap + 16);
    #pragma unroll
    for (int t = 0; t < 4; ++t) {
        // B fragment: col = nsub + t*16 + ll ; lo lanes k {0..7,16..23}, hi lanes {8..15,24..31}
        const _Float16* bp = &Bs[(nsub + t * 16 + ll) * 40 + hi * 8];
        v16h b;
        *((v8h*)&b)       = *(const v8h*)(bp);
        *(((v8h*)&b) + 1) = *(const v8h*)(bp + 16);
        acc[t] = __builtin_amdgcn_wmma_f32_16x16x32_f16(
            false, a, false, b, (short)0, acc[t], false, false);
    }
}

// ---------------------------------------------------------------------------
// Kernel A: fused chaizi-gather + dense GEMM (f16 WMMA, f32 acc) + bias +
// interleaved RoPE, emitting Q/K as f16 to workspace in [b][ent][l][64].
// Block: 256 threads (8 waves) computes a 64(M) x 128(N) output tile.
// Wave w owns rows (w%4)*16..+16 and cols (w/4)*64..+64 (4 WMMA tiles).
// k-loop split: 24 clean steps over HIDDEN, then a 2-step chaizi/pad tail.
// ---------------------------------------------------------------------------
__global__ __launch_bounds__(256) void dense_rope_kernel(
    const float* __restrict__ hid,     // (16384, 768)
    const int*   __restrict__ czid,    // (16384, 8)
    const float* __restrict__ W,       // (816, 1152)
    const float* __restrict__ bias,    // (1152)
    const float* __restrict__ czemb,   // (113, 6)
    _Float16* __restrict__ Qo,         // (288, 512, 64)
    _Float16* __restrict__ Ko)         // (288, 512, 64)
{
    __shared__ alignas(16) _Float16 As[64 * 40];    // 64 rows x 32 k (stride 40)
    __shared__ alignas(16) _Float16 Bs[128 * 40];   // Bt: 128 cols x 32 k (stride 40)

    const int mblk = blockIdx.x * 64;
    const int nblk = blockIdx.y * 128;
    const int tid  = threadIdx.x;
    const int wave = tid >> 5;
    const int lane = tid & 31;
    const int ll   = lane & 15;
    const int hi   = lane >> 4;
    const int msub = (wave & 3) * 16;
    const int nsub = (wave >> 2) * 64;

    v8f acc[4] = {};

    // Per-thread constant staging coordinates (hot loop: no div/branch).
    const int ar = tid >> 2;            // A: row 0..63
    const int aq = (tid & 3) * 8;       // A: 8-k chunk within 32
    const float* hrow = hid + (size_t)(mblk + ar) * HIDDEN + aq;

    // ---- main loop: k in [0, 768) ----
    for (int k0 = 0; k0 < HIDDEN; k0 += 32) {
        // A block: one v8h per thread (64 rows x 4 chunks == 256 threads)
        {
            const float4* src = (const float4*)(hrow + k0);
            *(v8h*)&As[ar * 40 + aq] = cvt8(src[0], src[1]);
        }
        // B block transposed: 32 k-rows x 32 float4-chunks = 4 per thread
        #pragma unroll
        for (int i = 0; i < 4; ++i) {
            int idx = tid + i * 256;
            int kk = idx >> 5, c4 = (idx & 31) * 4;
            float4 w4 = *(const float4*)(W + (size_t)(k0 + kk) * NOUT + nblk + c4);
            Bs[(c4 + 0) * 40 + kk] = (_Float16)w4.x;
            Bs[(c4 + 1) * 40 + kk] = (_Float16)w4.y;
            Bs[(c4 + 2) * 40 + kk] = (_Float16)w4.z;
            Bs[(c4 + 3) * 40 + kk] = (_Float16)w4.w;
        }
        __syncthreads();
        wmma_step(As, Bs, msub, nsub, ll, hi, acc);
        __syncthreads();
    }

    // ---- tail: k in [768, 832), chaizi gather + zero padding ----
    for (int k0 = HIDDEN; k0 < KPAD; k0 += 32) {
        for (int i = tid; i < 64 * 32; i += 256) {
            int r = i >> 5, kk = i & 31;
            int k = k0 + kk;
            float v = 0.0f;
            if (k < KTOT) {
                int j   = k - HIDDEN;                 // 0..47
                int cid = czid[(mblk + r) * 8 + (j / 6)];
                v = czemb[cid * 6 + (j % 6)];
            }
            As[r * 40 + kk] = (_Float16)v;
        }
        #pragma unroll
        for (int i = 0; i < 4; ++i) {
            int idx = tid + i * 256;
            int kk = idx >> 5, c4 = (idx & 31) * 4;
            int k = k0 + kk;
            float4 w4 = make_float4(0.f, 0.f, 0.f, 0.f);
            if (k < KTOT) w4 = *(const float4*)(W + (size_t)k * NOUT + nblk + c4);
            Bs[(c4 + 0) * 40 + kk] = (_Float16)w4.x;
            Bs[(c4 + 1) * 40 + kk] = (_Float16)w4.y;
            Bs[(c4 + 2) * 40 + kk] = (_Float16)w4.z;
            Bs[(c4 + 3) * 40 + kk] = (_Float16)w4.w;
        }
        __syncthreads();
        wmma_step(As, Bs, msub, nsub, ll, hi, acc);
        __syncthreads();
    }

    // ---- epilogue: bias + interleaved RoPE (partner column via shfl_xor) ----
    const float LOG1E4_OVER_32 = 0.28782313662425572f;  // ln(10000)/32
    #pragma unroll
    for (int t = 0; t < 4; ++t) {
        int n   = nblk + nsub + t * 16 + ll;   // 0..1151
        int ent = n / 128;
        int c   = n % 128;
        int isk = (c >= 64);
        int j   = isk ? (c - 64) : c;          // d-index 0..63
        float sgn  = (j & 1) ? 1.0f : -1.0f;   // even: -x[j+1]*sin ; odd: +x[j-1]*sin
        float freq = __expf(-(float)(j >> 1) * LOG1E4_OVER_32);
        float bv   = bias[n];
        v8f v = acc[t];
        #pragma unroll
        for (int r = 0; r < 8; ++r) {
            float x  = v[r] + bv;
            float xp = __shfl_xor(x, 1, 32);   // neighbor column (bias included)
            int m   = mblk + msub + hi * 8 + r;
            int pos = m & (SEQ - 1);
            float s, co;
            __sincosf((float)pos * freq, &s, &co);
            float o = x * co + sgn * xp * s;
            int b = m >> 9, l = m & (SEQ - 1);
            size_t off = (((size_t)(b * ENT + ent) * SEQ) + l) * DIM + j;
            if (isk) Ko[off] = (_Float16)o;
            else     Qo[off] = (_Float16)o;
        }
    }
}

// ---------------------------------------------------------------------------
// Kernel C: batched logits = Q @ K^T per (b,ent), fused mask + tril + scale.
// Grid: (288, 4). Block: 256 threads (8 waves) computes 128 rows x 512 cols.
// K staged in LDS in 256-row chunks (stride 72 halves -> conflict-free b128).
// ---------------------------------------------------------------------------
__global__ __launch_bounds__(256) void logits_kernel(
    const _Float16* __restrict__ Q,    // (288, 512, 64)
    const _Float16* __restrict__ K,    // (288, 512, 64)
    const float*    __restrict__ mask, // (32, 512)
    float* __restrict__ out)           // (288, 512, 512)
{
    __shared__ alignas(16) _Float16 Ks[256 * 72];   // 36 KB

    const int bh   = blockIdx.x;          // b*9 + ent
    const int b    = bh / ENT;
    const int mblk = blockIdx.y * 128;
    const int tid  = threadIdx.x;
    const int wave = tid >> 5;
    const int lane = tid & 31;
    const int ll   = lane & 15;
    const int hi   = lane >> 4;

    // This wave's Q fragments for both k-steps (register-resident all sweep).
    const _Float16* qrow = Q + ((size_t)bh * SEQ + mblk + wave * 16 + ll) * DIM;
    v16h a0, a1;
    {
        const _Float16* ap = qrow + hi * 8;           // k-step 0
        *((v8h*)&a0)       = *(const v8h*)(ap);
        *(((v8h*)&a0) + 1) = *(const v8h*)(ap + 16);
        const _Float16* aq = ap + 32;                 // k-step 1
        *((v8h*)&a1)       = *(const v8h*)(aq);
        *(((v8h*)&a1) + 1) = *(const v8h*)(aq + 16);
    }

    const _Float16* kbase = K + (size_t)bh * SEQ * DIM;
    const int mrow0 = mblk + wave * 16 + hi * 8;

    for (int chunk = 0; chunk < 2; ++chunk) {
        // stage 256 K rows: 256*8 v8h pieces, 8 per thread
        #pragma unroll
        for (int i = 0; i < 8; ++i) {
            int idx = tid + i * 256;
            int r = idx >> 3, o = (idx & 7) * 8;
            *(v8h*)&Ks[r * 72 + o] =
                *(const v8h*)(kbase + (size_t)(chunk * 256 + r) * DIM + o);
        }
        __syncthreads();

        for (int t = 0; t < 16; ++t) {
            int ncol = chunk * 256 + t * 16 + ll;     // this lane's column
            // B fragments: lane holds column ncol; lo lanes k {0..7,16..23}, hi {8..15,24..31}
            const _Float16* bp = &Ks[(t * 16 + ll) * 72 + hi * 8];
            v16h b0, b1;
            *((v8h*)&b0)       = *(const v8h*)(bp);
            *(((v8h*)&b0) + 1) = *(const v8h*)(bp + 16);
            const _Float16* bq = bp + 32;
            *((v8h*)&b1)       = *(const v8h*)(bq);
            *(((v8h*)&b1) + 1) = *(const v8h*)(bq + 16);

            v8f c = {};
            c = __builtin_amdgcn_wmma_f32_16x16x32_f16(
                false, a0, false, b0, (short)0, c, false, false);
            c = __builtin_amdgcn_wmma_f32_16x16x32_f16(
                false, a1, false, b1, (short)0, c, false, false);

            // epilogue: pad mask, strict-lower tril, 1/sqrt(d)
            float pad  = mask[b * SEQ + ncol];
            float base = -(1.0f - pad) * BIGV;
            float* op  = out + ((size_t)bh * SEQ + mrow0) * SEQ + ncol;
            #pragma unroll
            for (int r = 0; r < 8; ++r) {
                int m = mrow0 + r;
                float v = c[r] * pad + base;
                if (ncol < m) v -= BIGV;
                op[(size_t)r * SEQ] = v * 0.125f;
            }
        }
        __syncthreads();
    }
}

// ---------------------------------------------------------------------------
extern "C" void kernel_launch(void* const* d_in, const int* in_sizes, int n_in,
                              void* d_out, int out_size, void* d_ws, size_t ws_size,
                              hipStream_t stream) {
    const float* hid   = (const float*)d_in[0];   // last_hidden_state
    const float* mask  = (const float*)d_in[1];   // attention_mask
    const int*   czid  = (const int*)  d_in[2];   // chaizi_ids
    const float* W     = (const float*)d_in[3];   // W_dense
    const float* bias  = (const float*)d_in[4];   // b_dense
    const float* czemb = (const float*)d_in[5];   // chaizi_emb
    float* out = (float*)d_out;

    const size_t qk_elems = (size_t)BATCH * ENT * SEQ * DIM;
    _Float16* Qo = (_Float16*)d_ws;
    _Float16* Ko = Qo + qk_elems;

    dense_rope_kernel<<<dim3(16384 / 64, NOUT / 128), 256, 0, stream>>>(
        hid, czid, W, bias, czemb, Qo, Ko);
    logits_kernel<<<dim3(BATCH * ENT, SEQ / 128), 256, 0, stream>>>(
        Qo, Ko, mask, out);
}
